// Block_3633542332806
// MI455X (gfx1250) — compile-verified, matrix operand order checked by reading, not run on hardware
//
#include <hip/hip_runtime.h>
#include <cstdint>
#include <cstddef>

// ---------------------------------------------------------------------------
// Types for CDNA5 WMMA (bf16 in, f32 accumulate)
// ---------------------------------------------------------------------------
typedef __bf16 bf16;
typedef __attribute__((ext_vector_type(8)))  __bf16 v8bf;
typedef __attribute__((ext_vector_type(16))) __bf16 v16bf;
typedef __attribute__((ext_vector_type(8)))  float  v8f;

static constexpr int S  = 4096;
static constexpr int E  = 1024;
static constexpr int H  = 16;
static constexpr int E3 = 3 * E;     // 3072
static constexpr int E4 = 4 * E;     // 4096
static constexpr float INV_SCALE = 0.03125f;   // 1/sqrt(E) = 1/32

// Build a 16-element bf16 fragment from two contiguous 8-element (16B) runs.
// CDNA5 16-bit A/B fragment layout: VGPR0..3 = K = hf*8+0..7,
// VGPR4..7 = K = 16+hf*8+0..7 (hf = lane>>4).
__device__ __forceinline__ v16bf frag16(const bf16* p0, const bf16* p1) {
  v8bf lo = *(const v8bf*)p0;
  v8bf hi = *(const v8bf*)p1;
  return __builtin_shufflevector(lo, hi, 0,1,2,3,4,5,6,7,8,9,10,11,12,13,14,15);
}

__device__ __forceinline__ v8f wmma_bf16(v16bf a, v16bf b, v8f c) {
  return __builtin_amdgcn_wmma_f32_16x16x32_bf16(
      false, a, false, b, (short)0, c, false, false);
}

// CDNA5 async copy: global -> LDS, 16 bytes per lane, tracked by ASYNCcnt.
// Async ops complete in order, so partial waits retire the oldest copies.
__device__ __forceinline__ void async_copy_b128(uint32_t lds_byte, const void* g) {
  asm volatile("global_load_async_to_lds_b128 %0, %1, off"
               :: "v"(lds_byte), "v"((uint64_t)(uintptr_t)g) : "memory");
}
__device__ __forceinline__ void wait_async_le0() {
  asm volatile("s_wait_asynccnt 0" ::: "memory");
}
__device__ __forceinline__ void wait_async_le2() {
  asm volatile("s_wait_asynccnt 2" ::: "memory");
}

__device__ __forceinline__ float gelu_tanh(float x) {
  float x3 = x * x * x;
  return 0.5f * x * (1.0f + tanhf(0.7978845608028654f * (x + 0.044715f * x3)));
}

// ---------------------------------------------------------------------------
// Weight convert + transpose: src f32 [K][N] row-major -> dst bf16 [N][K]
// ---------------------------------------------------------------------------
__global__ void wt_kernel(const float* __restrict__ src, bf16* __restrict__ dst,
                          int K, int N) {
  int idx = blockIdx.x * blockDim.x + threadIdx.x;
  if (idx >= N * K) return;
  int n = idx / K;
  int k = idx - n * K;
  dst[idx] = (bf16)src[(size_t)k * N + n];
}

// ---------------------------------------------------------------------------
// LayerNorm (fp32 in -> bf16 out), one block (256 thr) per row, E = 1024
// ---------------------------------------------------------------------------
__global__ __launch_bounds__(256) void ln_kernel(const float* __restrict__ x,
                                                 const float* __restrict__ g,
                                                 const float* __restrict__ b,
                                                 bf16* __restrict__ out) {
  __shared__ float s1[256], s2[256];
  const int row = blockIdx.x;
  const int t = threadIdx.x;
  const float* xr = x + (size_t)row * E;
  float v[4], sum = 0.f, sq = 0.f;
#pragma unroll
  for (int i = 0; i < 4; ++i) {
    v[i] = xr[t + i * 256];
    sum += v[i];
    sq  += v[i] * v[i];
  }
  s1[t] = sum; s2[t] = sq;
  __syncthreads();
#pragma unroll
  for (int off = 128; off > 0; off >>= 1) {
    if (t < off) { s1[t] += s1[t + off]; s2[t] += s2[t + off]; }
    __syncthreads();
  }
  const float mu  = s1[0] * (1.0f / E);
  const float var = s2[0] * (1.0f / E) - mu * mu;
  const float rs  = rsqrtf(var + 1e-5f);
  bf16* orow = out + (size_t)row * E;
#pragma unroll
  for (int i = 0; i < 4; ++i) {
    int c = t + i * 256;
    orow[c] = (bf16)((v[i] - mu) * rs * g[c] + b[c]);
  }
}

// ---------------------------------------------------------------------------
// GEMM: C[M][N] = A[M][K](bf16) @ Bt[N][K](bf16)^T + bias, optional GELU /
// residual(f32) / bf16-vs-f32 output. Block = 256 thr (8 waves), block tile
// 128(M) x 256(N), wave tile 64x64 = 4x4 WMMA tiles. K ping-pong in steps of
// 2x32 so fragment loads for one stage overlap WMMAs of the other.
// ---------------------------------------------------------------------------
template <bool GELU, bool RES, bool OUTBF>
__global__ __launch_bounds__(256) void gemm_kernel(
    const bf16* __restrict__ A, const bf16* __restrict__ Bt,
    const float* __restrict__ bias, const float* __restrict__ res,
    float* __restrict__ outf, bf16* __restrict__ outb,
    int M, int N, int K) {
  const int t    = threadIdx.x;
  const int wave = t >> 5;
  const int lane = t & 31;
  const int hf   = lane >> 4;
  const int r16  = lane & 15;

  const int m0 = blockIdx.y * 128 + (wave >> 2) * 64;
  const int n0 = blockIdx.x * 256 + (wave & 3) * 64;

  v8f c[4][4] = {};

  const bf16* arow[4];
  const bf16* brow[4];
#pragma unroll
  for (int i = 0; i < 4; ++i)
    arow[i] = A + (size_t)(m0 + i * 16 + r16) * K + hf * 8;
#pragma unroll
  for (int j = 0; j < 4; ++j)
    brow[j] = Bt + (size_t)(n0 + j * 16 + r16) * K + hf * 8;

  v16bf a0[4], b0[4], a1[4], b1[4];

  auto loadf = [&](v16bf* a, v16bf* b, int kk) {
#pragma unroll
    for (int i = 0; i < 4; ++i) a[i] = frag16(arow[i] + kk, arow[i] + kk + 16);
#pragma unroll
    for (int j = 0; j < 4; ++j) b[j] = frag16(brow[j] + kk, brow[j] + kk + 16);
  };
  auto mmall = [&](v16bf* a, v16bf* b) {
#pragma unroll
    for (int i = 0; i < 4; ++i)
#pragma unroll
      for (int j = 0; j < 4; ++j)
        c[i][j] = wmma_bf16(a[i], b[j], c[i][j]);
  };

  loadf(a0, b0, 0);
  for (int k0 = 0; k0 < K; k0 += 64) {
    loadf(a1, b1, k0 + 32);
    if (k0 + 128 < K) {                         // keep L2->L0 stream warm
      __builtin_prefetch(arow[0] + k0 + 128, 0, 1);
      __builtin_prefetch(brow[0] + k0 + 128, 0, 1);
    }
    mmall(a0, b0);
    if (k0 + 64 < K) loadf(a0, b0, k0 + 64);
    mmall(a1, b1);
  }

  // Epilogue. C-frag layout: col = n0+j*16+r16 ; rows = m0+i*16+hf*8+r.
#pragma unroll
  for (int j = 0; j < 4; ++j) {
    const int col = n0 + j * 16 + r16;
    const float bv = bias[col];
#pragma unroll
    for (int i = 0; i < 4; ++i) {
      const int mb = m0 + i * 16 + hf * 8;
#pragma unroll
      for (int r = 0; r < 8; ++r) {
        float val = c[i][j][r] + bv;
        if (GELU) val = gelu_tanh(val);
        if (RES)  val += res[(size_t)(mb + r) * N + col];
        if (OUTBF) outb[(size_t)(mb + r) * N + col] = (bf16)val;
        else       outf[(size_t)(mb + r) * N + col] = val;
      }
    }
  }
}

// ---------------------------------------------------------------------------
// V transpose: qkv bf16 [S][3E] -> Vt bf16 [h*64+d][S]
// ---------------------------------------------------------------------------
__global__ void vt_kernel(const bf16* __restrict__ qkv, bf16* __restrict__ vt) {
  int idx = blockIdx.x * blockDim.x + threadIdx.x;
  if (idx >= E * S) return;
  int hd = idx / S;
  int s  = idx - hd * S;
  int h  = hd >> 6;
  int d  = hd & 63;
  vt[idx] = qkv[(size_t)s * E3 + h * 192 + 128 + d];
}

// ---------------------------------------------------------------------------
// Flash attention. Block = (head, 128-query tile), 256 thr = 8 waves.
// Double-buffered async K tiles + single V tile staged via CDNA5 async
// global->LDS copies; in-order ASYNCcnt lets partial waits overlap the V
// transfer with scores+softmax and the next K transfer with the whole tail
// of the current iteration. Probs (bf16) overlay the f32 score buffer.
// ---------------------------------------------------------------------------
__global__ __launch_bounds__(256) void attn_kernel(const bf16* __restrict__ qkv,
                                                   const bf16* __restrict__ vt,
                                                   bf16* __restrict__ ctx) {
  __shared__ float Sl[128 * 64];            // scores f32 / probs bf16 overlay
  __shared__ bf16  Kl[2][64 * 64];          // [key][d], double buffered
  __shared__ bf16  Vl[64 * 64];             // [d][key]
  __shared__ float rmax[128], rsum[128], ralpha[128];
  bf16* Pl = (bf16*)Sl;                     // probs: row stride 128 bf16

  const int h    = blockIdx.y;
  const int q0   = blockIdx.x * 128;
  const int t    = threadIdx.x;
  const int wave = t >> 5;
  const int lane = t & 31;
  const int hf   = lane >> 4;
  const int r16  = lane & 15;

  // 512 16B chunks per 64x64 bf16 tile; 2 per thread
  const int row_c = t >> 3;                 // via ci = t + i*256: rows below
  const int col_c = (t & 7) * 8;
  (void)row_c; (void)col_c;

  auto stage_K = [&](int kbase, bf16* dst) {
#pragma unroll
    for (int i = 0; i < 2; ++i) {
      const int ci  = t + i * 256;
      const int row = ci >> 3;
      const int col = (ci & 7) * 8;
      async_copy_b128((uint32_t)(uintptr_t)&dst[row * 64 + col],
                      qkv + (size_t)(kbase + row) * E3 + h * 192 + 64 + col);
    }
  };
  auto stage_V = [&](int kbase) {
#pragma unroll
    for (int i = 0; i < 2; ++i) {
      const int ci  = t + i * 256;
      const int row = ci >> 3;
      const int col = (ci & 7) * 8;
      async_copy_b128((uint32_t)(uintptr_t)&Vl[row * 64 + col],
                      vt + (size_t)(h * 64 + row) * S + kbase + col);
    }
  };

  if (t < 128) { rmax[t] = -1e30f; rsum[t] = 0.f; }

  v8f acc[4] = {};   // 16 queries x 64 d per wave (4 tiles along d)

  // Loop-invariant Q fragments (d = 0..31 and 32..63)
  const bf16* qrow = qkv + (size_t)(q0 + wave * 16 + r16) * E3 + h * 192 + hf * 8;
  const v16bf aq0 = frag16(qrow,      qrow + 16);
  const v16bf aq1 = frag16(qrow + 32, qrow + 48);

  const int kend = q0 + 128;
  stage_K(0, Kl[0]);                         // prefetch first K tile

  for (int kb = 0; kb < kend; kb += 64) {
    const int buf = (kb >> 6) & 1;
    __syncthreads();                         // Vl + prob buffer reusable
    stage_V(kb);                             // overlaps scores + softmax

    wait_async_le2();                        // K tile (oldest 2) has landed
    __syncthreads();

    // ---- scores: 16x64 tile per wave, S = Q @ K^T over d=64 ----
    const bf16* Kcur = Kl[buf];
    v8f sf[4] = {};
#pragma unroll
    for (int j = 0; j < 4; ++j) {
      const bf16* kr = &Kcur[(j * 16 + r16) * 64 + hf * 8];
      sf[j] = wmma_bf16(aq0, frag16(kr, kr + 16), sf[j]);
      sf[j] = wmma_bf16(aq1, frag16(kr + 32, kr + 48), sf[j]);
    }

    // prefetch next K tile into the other buffer (last read 1 iter ago)
    if (kb + 64 < kend) stage_K(kb + 64, Kl[buf ^ 1]);

#pragma unroll
    for (int j = 0; j < 4; ++j)
#pragma unroll
      for (int r = 0; r < 8; ++r)
        Sl[(wave * 16 + hf * 8 + r) * 64 + j * 16 + r16] = sf[j][r];
    __syncthreads();

    // ---- online softmax, one thread per query row ----
    if (t < 128) {
      const int srow = q0 + t;
      float vals[64];
      const float mold = rmax[t];
      float mnew = mold;
#pragma unroll
      for (int j = 0; j < 64; ++j) {
        float v = Sl[t * 64 + j] * INV_SCALE;
        if (kb + j > srow) v = -1e30f;
        vals[j] = v;
        mnew = fmaxf(mnew, v);
      }
      const float alpha = __expf(mold - mnew);
      float l = rsum[t] * alpha;
#pragma unroll
      for (int j = 0; j < 64; ++j) {
        float p = __expf(vals[j] - mnew);
        l += p;
        Pl[t * 128 + j] = (bf16)p;   // in-place overlay; vals[] already read
      }
      rmax[t] = mnew; rsum[t] = l; ralpha[t] = alpha;
    }

    wait_async_le2();                // V tile done (oldest); K-next may fly
    __syncthreads();

    // ---- rescale running ctx by alpha (per query row) ----
#pragma unroll
    for (int r = 0; r < 8; ++r) {
      const float a_ = ralpha[wave * 16 + hf * 8 + r];
#pragma unroll
      for (int d = 0; d < 4; ++d) acc[d][r] *= a_;
    }

    // ---- ctx += P @ V  (K-dim = 64 keys, both operands in LDS) ----
    const bf16* prow = &Pl[(wave * 16 + r16) * 128 + hf * 8];
    const v16bf ap0 = frag16(prow,      prow + 16);
    const v16bf ap1 = frag16(prow + 32, prow + 48);
#pragma unroll
    for (int d = 0; d < 4; ++d) {
      const bf16* vr = &Vl[(d * 16 + r16) * 64 + hf * 8];
      acc[d] = wmma_bf16(ap0, frag16(vr, vr + 16), acc[d]);
      acc[d] = wmma_bf16(ap1, frag16(vr + 32, vr + 48), acc[d]);
    }
  }

  wait_async_le0();
  __syncthreads();
  // ---- normalize and store ctx (bf16, [S][E] head-major columns) ----
#pragma unroll
  for (int r = 0; r < 8; ++r) {
    const int row = q0 + wave * 16 + hf * 8 + r;
    const float inv = 1.0f / rsum[wave * 16 + hf * 8 + r];
#pragma unroll
    for (int d = 0; d < 4; ++d)
      ctx[(size_t)row * E + h * 64 + d * 16 + r16] = (bf16)(acc[d][r] * inv);
  }
}

// ---------------------------------------------------------------------------
// Host launcher
// ---------------------------------------------------------------------------
extern "C" void kernel_launch(void* const* d_in, const int* in_sizes, int n_in,
                              void* d_out, int out_size, void* d_ws, size_t ws_size,
                              hipStream_t stream) {
  const float* hidden = (const float*)d_in[0];
  const float* ln1_g  = (const float*)d_in[1];
  const float* ln1_b  = (const float*)d_in[2];
  const float* qkv_w  = (const float*)d_in[3];
  const float* qkv_b  = (const float*)d_in[4];
  const float* out_w  = (const float*)d_in[5];
  const float* out_b  = (const float*)d_in[6];
  const float* ln2_g  = (const float*)d_in[7];
  const float* ln2_b  = (const float*)d_in[8];
  const float* up_w   = (const float*)d_in[9];
  const float* up_b   = (const float*)d_in[10];
  const float* down_w = (const float*)d_in[11];
  const float* down_b = (const float*)d_in[12];
  float* out = (float*)d_out;

  char* p = (char*)d_ws;
  auto alloc = [&](size_t bytes) -> char* {
    char* r = p;
    p += (bytes + 255) & ~(size_t)255;
    return r;
  };
  bf16* h1     = (bf16*)alloc((size_t)S * E * 2);
  bf16* qkvwt  = (bf16*)alloc((size_t)E3 * E * 2);
  bf16* qkvb   = (bf16*)alloc((size_t)S * E3 * 2);
  bf16* vtb    = (bf16*)alloc((size_t)E * S * 2);
  bf16* ctxb   = (bf16*)alloc((size_t)S * E * 2);
  bf16* outwt  = (bf16*)alloc((size_t)E * E * 2);
  float* xbuf  = (float*)alloc((size_t)S * E * 4);
  bf16* h2     = (bf16*)alloc((size_t)S * E * 2);
  bf16* upwt   = (bf16*)alloc((size_t)E4 * E * 2);
  bf16* upb    = (bf16*)alloc((size_t)S * E4 * 2);
  bf16* downwt = (bf16*)alloc((size_t)E * E4 * 2);

  dim3 blk(256);

  wt_kernel<<<(E * E3 + 255) / 256, blk, 0, stream>>>(qkv_w, qkvwt, E, E3);
  wt_kernel<<<(E * E  + 255) / 256, blk, 0, stream>>>(out_w, outwt, E, E);
  wt_kernel<<<(E * E4 + 255) / 256, blk, 0, stream>>>(up_w,  upwt,  E, E4);
  wt_kernel<<<(E4 * E + 255) / 256, blk, 0, stream>>>(down_w, downwt, E4, E);

  ln_kernel<<<S, blk, 0, stream>>>(hidden, ln1_g, ln1_b, h1);

  gemm_kernel<false, false, true><<<dim3(E3 / 256, S / 128), blk, 0, stream>>>(
      h1, qkvwt, qkv_b, nullptr, nullptr, qkvb, S, E3, E);

  vt_kernel<<<(E * S + 255) / 256, blk, 0, stream>>>(qkvb, vtb);

  attn_kernel<<<dim3(S / 128, H), blk, 0, stream>>>(qkvb, vtb, ctxb);

  gemm_kernel<false, true, false><<<dim3(E / 256, S / 128), blk, 0, stream>>>(
      ctxb, outwt, out_b, hidden, xbuf, nullptr, S, E, E);

  ln_kernel<<<S, blk, 0, stream>>>(xbuf, ln2_g, ln2_b, h2);

  gemm_kernel<true, false, true><<<dim3(E4 / 256, S / 128), blk, 0, stream>>>(
      h2, upwt, up_b, nullptr, nullptr, upb, S, E4, E);

  gemm_kernel<false, true, false><<<dim3(E / 256, S / 128), blk, 0, stream>>>(
      upb, downwt, down_b, xbuf, out, nullptr, S, E, E4);
}
